// INR2D_27736898797741
// MI455X (gfx1250) — compile-verified, matrix operand order checked by reading, not run on hardware
//
#include <hip/hip_runtime.h>
#include <hip/hip_fp16.h>

typedef __attribute__((ext_vector_type(16))) _Float16 v16h;
typedef __attribute__((ext_vector_type(8)))  _Float16 v8h;
typedef __attribute__((ext_vector_type(8)))  float    v8f;

#define BB   4
#define CC   64
#define HH   128
#define WW   128
#define TH   256
#define TW   256
#define NPTS (TH*TW)
#define ROWS 128
#define XS   104   // x-buffer row stride (halfs), K padded 96 + 8
#define AS   264   // activation row stride (halfs), 256 + 8 (bank-conflict pad)

// Force a global (addrspace 1) vector load even when the pointer's provenance
// has been made opaque: emits global_load_b128 (LOADcnt) instead of
// flat_load_b128 (LOADcnt+DScnt, ties up the LDS pipe).
#define GLOAD8H(p) (*(const __attribute__((address_space(1))) v8h*)(p))

union Frag {
  v16h v;
  struct { v8h lo, hi; } h;
};

__device__ __forceinline__ v8f wmma_f16(const Frag& a, const Frag& b, v8f c) {
  return __builtin_amdgcn_wmma_f32_16x16x32_f16(false, a.v, false, b.v,
                                                (short)0, c, false, false);
}

// Per-point bilinear neighbor parameters (corner coords/weights + rel offsets).
__device__ __forceinline__ void neigh(float cy, float cx, int vx, int vy,
                                      int& x0, int& x1, int& y0, int& y1,
                                      float& wx0, float& wx1, float& wy0, float& wy1,
                                      float& rel0, float& rel1)
{
  const float eps = 1e-6f;
  float c0 = cy + (float)vx * (1.0f / HH) + eps;
  float c1 = cx + (float)vy * (1.0f / WW) + eps;
  c0 = fminf(fmaxf(c0, -1.0f + 1e-6f), 1.0f - 1e-6f);
  c1 = fminf(fmaxf(c1, -1.0f + 1e-6f), 1.0f - 1e-6f);
  float ix = (c1 + 1.0f) * 0.5f * (float)(WW - 1);
  float iy = (c0 + 1.0f) * 0.5f * (float)(HH - 1);
  float x0f = floorf(ix), y0f = floorf(iy);
  x0 = min(max((int)x0f, 0), WW - 1);
  x1 = min(x0 + 1, WW - 1);
  y0 = min(max((int)y0f, 0), HH - 1);
  y1 = min(y0 + 1, HH - 1);
  wx1 = ix - x0f; wx0 = 1.0f - wx1;
  wy1 = iy - y0f; wy0 = 1.0f - wy1;
  float qy = wy0 * (-1.0f + (2.0f * y0 + 1.0f) / HH) + wy1 * (-1.0f + (2.0f * y1 + 1.0f) / HH);
  float qx = wx0 * (-1.0f + (2.0f * x0 + 1.0f) / WW) + wx1 * (-1.0f + (2.0f * x1 + 1.0f) / WW);
  rel0 = (cy - qy) * (float)HH;
  rel1 = (cx - qx) * (float)WW;
}

// N-split dense layer: this wave owns output columns [wid*32, wid*32+32) for
// ALL 128 rows. B fragments (2 n-tiles x KCH chunks) are pinned in VGPRs for
// the duration of the layer only (asm barrier defeats cross-layer hoisting).
// Per m-tile: preload all A fragments, then a 16-WMMA burst on 2 acc chains.
template <int KCH, bool RELU>
__device__ __forceinline__ void mlp_layer_n(const _Float16* __restrict__ actIn, int sIn,
                                            const _Float16* wT, int kpad,
                                            const float* __restrict__ bias,
                                            _Float16* __restrict__ actOut, int sOut,
                                            int lid, int wid)
{
  // Opaque-ify the weight pointer: keeps this layer's B loads from being
  // CSE'd/hoisted across layers or neighbor iterations (register pressure).
  asm volatile("" : "+s"(wT));

  const int n16 = lid & 15;
  const int hi8 = (lid >> 4) << 3;
  const int ncol = wid * 32;                 // first of 32 owned columns

  Frag B0[KCH], B1[KCH];
  const _Float16* w0r = wT + (ncol + n16) * kpad;
  const _Float16* w1r = w0r + 16 * kpad;
#pragma unroll
  for (int kc = 0; kc < KCH; ++kc) {
    int k0 = kc * 32 + hi8;
    B0[kc].h.lo = GLOAD8H(w0r + k0);
    B0[kc].h.hi = GLOAD8H(w0r + k0 + 16);
    B1[kc].h.lo = GLOAD8H(w1r + k0);
    B1[kc].h.hi = GLOAD8H(w1r + k0 + 16);
  }
  const float bv0 = bias[ncol + n16];
  const float bv1 = bias[ncol + 16 + n16];

#pragma unroll 2
  for (int mt = 0; mt < 8; ++mt) {
    const _Float16* arow = actIn + (mt * 16 + n16) * sIn;
    Frag A[KCH];
#pragma unroll
    for (int kc = 0; kc < KCH; ++kc) {          // issue all A loads up front
      int k0 = kc * 32 + hi8;
      A[kc].h.lo = *(const v8h*)(arow + k0);
      A[kc].h.hi = *(const v8h*)(arow + k0 + 16);
    }
    v8f acc0 = { bv0, bv0, bv0, bv0, bv0, bv0, bv0, bv0 };
    v8f acc1 = { bv1, bv1, bv1, bv1, bv1, bv1, bv1, bv1 };
#pragma unroll
    for (int kc = 0; kc < KCH; ++kc) {          // WMMA burst, 2 indep chains
      acc0 = wmma_f16(A[kc], B0[kc], acc0);
      acc1 = wmma_f16(A[kc], B1[kc], acc1);
    }
    _Float16* orow = actOut + (mt * 16 + hi8) * sOut + ncol + n16;
#pragma unroll
    for (int r = 0; r < 8; ++r) {
      float v0 = acc0[r], v1 = acc1[r];
      if (RELU) { v0 = fmaxf(v0, 0.0f); v1 = fmaxf(v1, 0.0f); }
      orow[r * sOut]      = (_Float16)v0;
      orow[r * sOut + 16] = (_Float16)v1;
    }
  }
}

// Transpose + f32->f16 convert weights: src[K][N] -> dst[Npad][Kpad] (zero padded)
__global__ void convT_kernel(const float* __restrict__ src, _Float16* __restrict__ dst,
                             int K, int Nn, int Kpad, int Npad)
{
  int i = blockIdx.x * blockDim.x + threadIdx.x;
  if (i >= Kpad * Npad) return;
  int n = i / Kpad;
  int k = i - n * Kpad;
  float v = (k < K && n < Nn) ? src[k * Nn + n] : 0.0f;
  dst[i] = (_Float16)v;
}

__global__ void __launch_bounds__(256)
inr_mlp_kernel(const float* __restrict__ feat,
               const float* __restrict__ b0, const float* __restrict__ b1,
               const float* __restrict__ b2, const float* __restrict__ b3,
               const _Float16* __restrict__ wT0, const _Float16* __restrict__ wT1,
               const _Float16* __restrict__ wT2, const _Float16* __restrict__ wT3,
               float* __restrict__ out)
{
  __shared__ _Float16 sA[ROWS * AS];
  __shared__ _Float16 sB[ROWS * AS];   // also hosts the assembled input (stride XS)
  __shared__ float sWX0[ROWS], sWX1[ROWS], sWY0[ROWS], sWY1[ROWS];
  __shared__ int   sIX0[ROWS], sIX1[ROWS], sIY0[ROWS], sIY1[ROWS];
  __shared__ float sWgt[ROWS], sTot[ROWS], sCY[ROWS], sCX[ROWS];

  const int tid = threadIdx.x;
  const int lid = tid & 31;
  const int wid = tid >> 5;
  const int n16 = lid & 15;
  const int hi8 = (lid >> 4) << 3;
  const int rowbase = wid * 16;
  const int base = blockIdx.x * ROWS;
  const int b = base / NPTS;            // 128 | NPTS, so whole block shares b

  // warm WGP$/L2 for this wave's weight slices (global_prefetch path)
  __builtin_prefetch(wT1 + (size_t)(wid * 32 + n16) * 256 + hi8, 0, 1);
  __builtin_prefetch(wT2 + (size_t)(wid * 32 + n16) * 256 + hi8, 0, 1);

  // ---- per-point invariants: coord + total area (for the cross-paired weights)
  if (tid < ROWS) {
    int pi = base + tid - b * NPTS;
    int y = pi / TW, x = pi - y * TW;
    float cy = -1.0f + (2.0f * y + 1.0f) / TH;
    float cx = -1.0f + (2.0f * x + 1.0f) / TW;
    sCY[tid] = cy; sCX[tid] = cx;
    float tot = 0.0f;
#pragma unroll
    for (int v = 0; v < 4; ++v) {
      int vx = (v < 2) ? -1 : 1, vy = (v & 1) ? 1 : -1;
      int a0, a1, a2, a3; float f0, f1, f2, f3, r0, r1;
      neigh(cy, cx, vx, vy, a0, a1, a2, a3, f0, f1, f2, f3, r0, r1);
      tot += fabsf(r0 * r1) + 1e-9f;
    }
    sTot[tid] = tot;
  }

  v8f outAcc = {};

  for (int v = 0; v < 4; ++v) {
    const int vx = (v < 2) ? -1 : 1, vy = (v & 1) ? 1 : -1;

    // ---- phase 1: per-point params + non-feature part of the input vector
    if (tid < ROWS) {
      float cy = sCY[tid], cx = sCX[tid];
      int x0, x1, y0, y1; float wx0, wx1, wy0, wy1, r0, r1;
      neigh(cy, cx, vx, vy, x0, x1, y0, y1, wx0, wx1, wy0, wy1, r0, r1);
      sIX0[tid] = x0; sIX1[tid] = x1; sIY0[tid] = y0; sIY1[tid] = y1;
      sWX0[tid] = wx0; sWX1[tid] = wx1; sWY0[tid] = wy0; sWY1[tid] = wy1;
      int t0, t1, t2, t3; float g0, g1, g2, g3, o0, o1;
      neigh(cy, cx, -vx, -vy, t0, t1, t2, t3, g0, g1, g2, g3, o0, o1);
      sWgt[tid] = (fabsf(o0 * o1) + 1e-9f) / sTot[tid];

      _Float16* xr = sB + tid * XS;
      xr[64] = (_Float16)r0;
      xr[65] = (_Float16)r1;
      float fr = 3.14159265358979f;
#pragma unroll
      for (int l = 0; l < 4; ++l) {
        xr[66 + l] = (_Float16)__sinf(cy * fr);
        xr[70 + l] = (_Float16)__cosf(cy * fr);
        xr[74 + l] = (_Float16)__sinf(cx * fr);
        xr[78 + l] = (_Float16)__cosf(cx * fr);
        fr *= 2.0f;
      }
      xr[82] = (_Float16)(2.0f * HH / TH);
      xr[83] = (_Float16)(2.0f * WW / TW);
#pragma unroll
      for (int k = 84; k < 96; ++k) xr[k] = (_Float16)0.0f;
    }
    __syncthreads();

    // ---- phase 2: gather 64 feature channels (2 threads/point, 32 ch each)
    {
      int p = tid >> 1;
      int c0 = (tid & 1) * 32;
      const int HWsz = HH * WW;
      float w00 = sWY0[p] * sWX0[p], w01 = sWY0[p] * sWX1[p];
      float w10 = sWY1[p] * sWX0[p], w11 = sWY1[p] * sWX1[p];
      const float* fb  = feat + (size_t)b * CC * HWsz;
      const float* p00 = fb + sIY0[p] * WW + sIX0[p];
      const float* p01 = fb + sIY0[p] * WW + sIX1[p];
      const float* p10 = fb + sIY1[p] * WW + sIX0[p];
      const float* p11 = fb + sIY1[p] * WW + sIX1[p];
      _Float16* xr = sB + p * XS + c0;
#pragma unroll 8
      for (int c = 0; c < 32; ++c) {
        int off = (c0 + c) * HWsz;
        float val = w00 * p00[off] + w01 * p01[off] + w10 * p10[off] + w11 * p11[off];
        xr[c] = (_Float16)val;
      }
    }
    __syncthreads();

    // ---- MLP: L0..L2 split the 256 output columns across the 8 waves
    mlp_layer_n<3, true>(sB, XS, wT0,  96, b0, sA, AS, lid, wid);
    __syncthreads();
    mlp_layer_n<8, true>(sA, AS, wT1, 256, b1, sB, AS, lid, wid);
    __syncthreads();
    mlp_layer_n<8, true>(sB, AS, wT2, 256, b2, sA, AS, lid, wid);
    __syncthreads();

    // ---- last layer (N=3 padded to 16): M-split, weighted accumulate in regs
    {
      const _Float16* wp = wT3;
      asm volatile("" : "+s"(wp));              // no cross-neighbor pinning
      float bv = (n16 < 3) ? b3[n16] : 0.0f;
      const _Float16* arow = sA + (rowbase + n16) * AS;
      const _Float16* wrow = wp + n16 * 256;
      Frag A[8], Bf[8];
#pragma unroll
      for (int kc = 0; kc < 8; ++kc) {
        int k0 = kc * 32 + hi8;
        A[kc].h.lo  = *(const v8h*)(arow + k0);
        A[kc].h.hi  = *(const v8h*)(arow + k0 + 16);
        Bf[kc].h.lo = GLOAD8H(wrow + k0);
        Bf[kc].h.hi = GLOAD8H(wrow + k0 + 16);
      }
      v8f acc = { bv, bv, bv, bv, bv, bv, bv, bv };
#pragma unroll
      for (int kc = 0; kc < 8; ++kc) acc = wmma_f16(A[kc], Bf[kc], acc);
#pragma unroll
      for (int r = 0; r < 8; ++r) {
        int m = rowbase + r + hi8;
        outAcc[r] += sWgt[m] * acc[r];
      }
    }
    __syncthreads();
  }

  // ---- write blended output: out[b][ch][y][x]
  if (n16 < 3) {
#pragma unroll
    for (int r = 0; r < 8; ++r) {
      int m = rowbase + r + hi8;
      int pi = base + m - b * NPTS;
      int y = pi / TW, x = pi - y * TW;
      out[((size_t)(b * 3 + n16) * TH + y) * TW + x] = outAcc[r];
    }
  }
}

extern "C" void kernel_launch(void* const* d_in, const int* in_sizes, int n_in,
                              void* d_out, int out_size, void* d_ws, size_t ws_size,
                              hipStream_t stream)
{
  const float* feat = (const float*)d_in[0];
  const float* w0   = (const float*)d_in[1];
  const float* b0   = (const float*)d_in[2];
  const float* w1   = (const float*)d_in[3];
  const float* b1   = (const float*)d_in[4];
  const float* w2   = (const float*)d_in[5];
  const float* b2   = (const float*)d_in[6];
  const float* w3   = (const float*)d_in[7];
  const float* b3   = (const float*)d_in[8];
  float* out = (float*)d_out;

  // Workspace: f16 transposed weights [N][Kpad]
  _Float16* wT0 = (_Float16*)d_ws;        // 256 x 96
  _Float16* wT1 = wT0 + 256 * 96;         // 256 x 256
  _Float16* wT2 = wT1 + 256 * 256;        // 256 x 256
  _Float16* wT3 = wT2 + 256 * 256;        // 16  x 256

  convT_kernel<<<(256 *  96 + 255) / 256, 256, 0, stream>>>(w0, wT0,  84, 256,  96, 256);
  convT_kernel<<<(256 * 256 + 255) / 256, 256, 0, stream>>>(w1, wT1, 256, 256, 256, 256);
  convT_kernel<<<(256 * 256 + 255) / 256, 256, 0, stream>>>(w2, wT2, 256, 256, 256, 256);
  convT_kernel<<<( 16 * 256 + 255) / 256, 256, 0, stream>>>(w3, wT3, 256,   3, 256,  16);

  const int grid = (BB * NPTS) / ROWS;    // 2048 workgroups of 8 waves
  inr_mlp_kernel<<<grid, 256, 0, stream>>>(feat, b0, b1, b2, b3,
                                           wT0, wT1, wT2, wT3, out);
}
